// QRobertaSelfAttention_88862873354637
// MI455X (gfx1250) — compile-verified, matrix operand order checked by reading, not run on hardware
//
#include <hip/hip_runtime.h>

typedef __attribute__((ext_vector_type(16))) _Float16 v16h;
typedef __attribute__((ext_vector_type(8)))  _Float16 v8h;
typedef __attribute__((ext_vector_type(8)))  float    v8f;

// Problem constants
#define BB_   4
#define SS_   2048
#define HID_  768
#define NH_   12
#define HD_   64

// Load 16 halves in WMMA A/B per-lane pattern: 8 at p, 8 at p+16.
__device__ __forceinline__ v16h ld16h(const _Float16* p) {
    v8h x0 = *(const v8h*)p;
    v8h x1 = *(const v8h*)(p + 16);
    return __builtin_shufflevector(x0, x1, 0,1,2,3,4,5,6,7,8,9,10,11,12,13,14,15);
}

// Same pattern but from f32 memory, converting to f16.
__device__ __forceinline__ v16h ld16f(const float* p) {
    v16h r;
#pragma unroll
    for (int i = 0; i < 8; ++i) r[i] = (_Float16)p[i];
#pragma unroll
    for (int i = 0; i < 8; ++i) r[8 + i] = (_Float16)p[16 + i];
    return r;
}

__device__ __forceinline__ v8f wmma_f16(v16h a, v16h b, v8f c) {
    return __builtin_amdgcn_wmma_f32_16x16x32_f16(false, a, false, b, (short)0, c, false, false);
}

// ---------------------------------------------------------------------------
// Kernel 0: W [K,N] f32 -> Wt [N,K] f16 (columns contiguous for WMMA B operand)
// ---------------------------------------------------------------------------
__global__ __launch_bounds__(256) void wtrans_kernel(const float* __restrict__ W,
                                                     _Float16* __restrict__ Wt) {
    int idx = blockIdx.x * 256 + threadIdx.x;
    if (idx >= HID_ * HID_) return;
    int n = idx / HID_, k = idx % HID_;
    Wt[(size_t)n * HID_ + k] = (_Float16)W[(size_t)k * HID_ + n];
}

// ---------------------------------------------------------------------------
// Kernel 1: out_codes = clamp(floor((hs@W + b)/s + 0.5), -16, 15)  stored as f16
// One wave computes a 16x64 output strip (4 N-tiles): the fp32->f16 converted
// A operand is reused by 4 WMMAs per K-step.
// vmode==0 : store [b,h,s,d]   (Q, K)
// vmode==1 : store [b,h,d,s]   (V transposed, so PV B-operand is t-contiguous)
// ---------------------------------------------------------------------------
__global__ __launch_bounds__(256) void qproj_kernel(
    const float* __restrict__ hs, const _Float16* __restrict__ Wt,
    const float* __restrict__ bias, const float* __restrict__ s_ptr,
    _Float16* __restrict__ outp, int vmode)
{
    const int wid    = threadIdx.x >> 5;
    const int lane   = threadIdx.x & 31;
    const int lrow   = lane & 15;
    const int hi     = lane >> 4;
    const int k0     = hi ? 8 : 0;
    const int tile   = blockIdx.x * 8 + wid;       // 0..6143
    const int tile_m = tile / (HID_ / 64);         // token tile 0..511
    const int ng     = tile % (HID_ / 64);         // 64-wide feature group 0..11

    const float* ap = hs + (size_t)(tile_m * 16 + lrow) * HID_ + k0;
    const _Float16* bp[4];
#pragma unroll
    for (int nt = 0; nt < 4; ++nt)
        bp[nt] = Wt + (size_t)(ng * 64 + nt * 16 + lrow) * HID_ + k0;

    v8f acc[4];
#pragma unroll
    for (int nt = 0; nt < 4; ++nt) acc[nt] = (v8f){};

    for (int kb = 0; kb < HID_ / 32; ++kb) {
        v16h a = ld16f(ap + kb * 32);              // one cvt'd A feeds 4 WMMAs
#pragma unroll
        for (int nt = 0; nt < 4; ++nt) {
            v16h b = ld16h(bp[nt] + kb * 32);
            acc[nt] = wmma_f16(a, b, acc[nt]);
        }
    }

    const float inv_s = 1.0f / (*s_ptr);
#pragma unroll
    for (int nt = 0; nt < 4; ++nt) {
        const int feat = ng * 64 + nt * 16 + lrow;
        const float bval = bias[feat];
        const int hh = feat >> 6, dd = feat & (HD_ - 1);
#pragma unroll
        for (int r = 0; r < 8; ++r) {
            const int tok = tile_m * 16 + r + hi * 8;
            const int bb  = tok >> 11, ss = tok & (SS_ - 1);
            float qv = fminf(fmaxf(floorf((acc[nt][r] + bval) * inv_s + 0.5f), -16.0f), 15.0f);
            size_t idx = vmode ? ((((size_t)bb * NH_ + hh) * HD_ + dd) * SS_ + ss)
                               : ((((size_t)bb * NH_ + hh) * SS_ + ss) * HD_ + dd);
            outp[idx] = (_Float16)qv;
        }
    }
}

// ---------------------------------------------------------------------------
// Kernel 2: attention. 8 waves/block, all waves share one (b,h) head, each
// wave owns one 16-query tile. K/V chunks are staged into LDS once per block
// (cooperative copy by all 256 threads) and consumed by all 8 waves via
// ds_load_b128 -- 8x reduction in global K/V traffic. Next chunk is
// global_prefetch'd while the current one is consumed.
//   pass 1: QK^T via WMMA; per-lane online softmax, one cross-lane merge.
//   pass 2: recompute scores, quantize probs (int codes in f16), LDS bounce
//           D-layout -> A-layout, WMMA with LDS-staged V^T.
// ---------------------------------------------------------------------------
__global__ __launch_bounds__(256) void qattn_kernel(
    const _Float16* __restrict__ qh, const _Float16* __restrict__ kh,
    const _Float16* __restrict__ vT, const float* __restrict__ mask,
    const float* __restrict__ sq_p, const float* __restrict__ sk_p,
    const float* __restrict__ sv_p, const float* __restrict__ sa_p,
    const float* __restrict__ saf_p, float* __restrict__ out)
{
    __shared__ __align__(16) _Float16 stage[4096];     // 8KB: pass1 K / pass2 K+V
    __shared__ __align__(16) _Float16 lds_p[8][16 * 32]; // 8KB: per-wave prob tiles

    const int tid  = threadIdx.x;
    const int wid  = tid >> 5;
    const int lane = tid & 31;
    const int lrow = lane & 15;
    const int hi   = lane >> 4;
    const int k0   = hi ? 8 : 0;

    const int tile  = blockIdx.x * 8 + wid;        // 0..6143
    const int stile = tile & 127;                  // query tile within sequence
    const int h     = (tile >> 7) % NH_;           // uniform across the block
    const int b     = tile / (128 * NH_);          // uniform across the block

    const size_t    headoff = ((size_t)b * NH_ + h) * (size_t)SS_ * HD_;
    const _Float16* qp  = qh + headoff + (size_t)(stile * 16 + lrow) * HD_;
    const _Float16* kb_ = kh + headoff;            // [t][d] contiguous
    const _Float16* vb_ = vT + headoff;            // [d][t]
    const float*    mb  = mask + (size_t)b * SS_;

    v16h a0 = ld16h(qp + k0);                      // q rows, d 0..31
    v16h a1 = ld16h(qp + 32 + k0);                 // q rows, d 32..63

    const float scale = (*sq_p) * (*sk_p) * 0.125f;  // 1/sqrt(64)

    float M[8], Z[8];
#pragma unroll
    for (int r = 0; r < 8; ++r) { M[r] = -3.0e38f; Z[r] = 0.0f; }

    // ---- Pass 1: stage 64 K-rows (8KB, contiguous) per block iteration ----
    for (int it = 0; it < SS_ / 64; ++it) {
        const _Float16* src = kb_ + (size_t)it * 4096;
        {
            const int i0 = tid * 16;               // contiguous block memcpy
            *(v8h*)(stage + i0)     = *(const v8h*)(src + i0);
            *(v8h*)(stage + i0 + 8) = *(const v8h*)(src + i0 + 8);
            if (it + 1 < SS_ / 64)
                __builtin_prefetch(src + 4096 + i0, 0, 3);  // global_prefetch_b8
        }
        __syncthreads();
#pragma unroll
        for (int sub = 0; sub < 4; ++sub) {
            const _Float16* kp = stage + (sub * 16 + lrow) * HD_ + k0;
            v16h b0 = ld16h(kp);
            v16h b1 = ld16h(kp + 32);
            v8f c = {};
            c = wmma_f16(a0, b0, c);
            c = wmma_f16(a1, b1, c);
            const float mv = mb[it * 64 + sub * 16 + lrow];
#pragma unroll
            for (int r = 0; r < 8; ++r) {
                float x  = c[r] * scale + mv;
                float nM = fmaxf(M[r], x);
                Z[r] = Z[r] * __expf(M[r] - nM) + __expf(x - nM);
                M[r] = nM;
            }
        }
        __syncthreads();
    }

    // ---- Cross-lane (M, Z) merge within each 16-lane row group -----------
#pragma unroll
    for (int r = 0; r < 8; ++r) {
#pragma unroll
        for (int d = 1; d < 16; d <<= 1) {
            float oM = __shfl_xor(M[r], d);
            float oZ = __shfl_xor(Z[r], d);
            float nM = fmaxf(M[r], oM);
            Z[r] = Z[r] * __expf(M[r] - nM) + oZ * __expf(oM - nM);
            M[r] = nM;
        }
    }

    float invZ[8];
#pragma unroll
    for (int r = 0; r < 8; ++r) invZ[r] = 1.0f / Z[r];

    // ---- Pass 2: stage 32 K-rows + 32 V^T-cols per iteration --------------
    const float inv_sa = 1.0f / (*sa_p);
    v8f acc[4];
#pragma unroll
    for (int dt = 0; dt < 4; ++dt) acc[dt] = (v8f){};
    _Float16* K2  = stage;                          // 32 x 64 halves (4KB)
    _Float16* V2  = stage + 2048;                   // 64 x 32 halves (4KB)
    _Float16* myp = lds_p[wid];

    const int krow = tid >> 3, kseg = tid & 7;      // K2 copy mapping
    const int vd   = tid >> 2, vseg = tid & 3;      // V2 copy mapping

    for (int tt2 = 0; tt2 < SS_ / 32; ++tt2) {
        const int tbase = tt2 * 32;
        {
            *(v8h*)(K2 + krow * HD_ + kseg * 8) =
                *(const v8h*)(kb_ + (size_t)(tbase + krow) * HD_ + kseg * 8);
            *(v8h*)(V2 + vd * 32 + vseg * 8) =
                *(const v8h*)(vb_ + (size_t)vd * SS_ + tbase + vseg * 8);
            if (tt2 + 1 < SS_ / 32) {
                __builtin_prefetch(kb_ + (size_t)(tbase + 32 + krow) * HD_ + kseg * 8, 0, 3);
                __builtin_prefetch(vb_ + (size_t)vd * SS_ + tbase + 32 + vseg * 8, 0, 3);
            }
        }
        __syncthreads();
#pragma unroll
        for (int sub = 0; sub < 2; ++sub) {
            const _Float16* kp = K2 + (sub * 16 + lrow) * HD_ + k0;
            v16h b0 = ld16h(kp);
            v16h b1 = ld16h(kp + 32);
            v8f c = {};
            c = wmma_f16(a0, b0, c);
            c = wmma_f16(a1, b1, c);
            const float mv = mb[tbase + sub * 16 + lrow];
#pragma unroll
            for (int r = 0; r < 8; ++r) {
                float x  = c[r] * scale + mv;
                float p  = __expf(x - M[r]) * invZ[r];
                float pi = fminf(fmaxf(floorf(p * inv_sa + 0.5f), 0.0f), 31.0f);
                myp[(r + hi * 8) * 32 + sub * 16 + lrow] = (_Float16)pi;
            }
        }
        // Intra-wave D-layout -> A-layout bounce: DS ops are in-order per
        // wave; a scheduling fence suffices (prob tiles are wave-private).
        __builtin_amdgcn_wave_barrier();
        v16h pa = ld16h(myp + lrow * 32 + k0);
        __builtin_amdgcn_wave_barrier();
#pragma unroll
        for (int dt = 0; dt < 4; ++dt) {
            v16h vv = ld16h(V2 + (dt * 16 + lrow) * 32 + k0);
            acc[dt] = wmma_f16(pa, vv, acc[dt]);
        }
        __syncthreads();
    }

    // ---- Epilogue: ctx scale + asymmetric fake-quant, write [B,S,HID] ----
    const float sfac    = (*sa_p) * (*sv_p);
    const float saf     = *saf_p;
    const float inv_saf = 1.0f / saf;
#pragma unroll
    for (int dt = 0; dt < 4; ++dt) {
        const int feat = h * HD_ + dt * 16 + lrow;
#pragma unroll
        for (int r = 0; r < 8; ++r) {
            const int sg   = stile * 16 + r + hi * 8;
            float ctxr = acc[dt][r] * sfac;
            float o = fminf(fmaxf(floorf(ctxr * inv_saf + 0.5f), 0.0f), 31.0f) * saf;
            out[((size_t)b * SS_ + sg) * HID_ + feat] = o;
        }
    }
}

// ---------------------------------------------------------------------------
extern "C" void kernel_launch(void* const* d_in, const int* in_sizes, int n_in,
                              void* d_out, int out_size, void* d_ws, size_t ws_size,
                              hipStream_t stream) {
    const float* hs      = (const float*)d_in[0];
    const float* mask    = (const float*)d_in[1];
    const float* Wq      = (const float*)d_in[2];
    const float* bq      = (const float*)d_in[3];
    const float* Wk      = (const float*)d_in[4];
    const float* bk      = (const float*)d_in[5];
    const float* Wv      = (const float*)d_in[6];
    const float* bv      = (const float*)d_in[7];
    const float* s_q     = (const float*)d_in[8];
    const float* s_k     = (const float*)d_in[9];
    const float* s_v     = (const float*)d_in[10];
    const float* s_attn  = (const float*)d_in[11];
    const float* s_after = (const float*)d_in[12];

    char* ws = (char*)d_ws;
    const size_t wbytes = (size_t)HID_ * HID_ * sizeof(_Float16);           // 1.18 MB
    const size_t qbytes = (size_t)BB_ * NH_ * SS_ * HD_ * sizeof(_Float16); // 12.6 MB
    _Float16* WtQ = (_Float16*)(ws + 0 * wbytes);
    _Float16* WtK = (_Float16*)(ws + 1 * wbytes);
    _Float16* WtV = (_Float16*)(ws + 2 * wbytes);
    _Float16* Qh  = (_Float16*)(ws + 3 * wbytes);
    _Float16* Kh  = (_Float16*)(ws + 3 * wbytes + qbytes);
    _Float16* VT  = (_Float16*)(ws + 3 * wbytes + 2 * qbytes);

    // Weight transpose + f16 convert
    const int wtot = HID_ * HID_;
    wtrans_kernel<<<(wtot + 255) / 256, 256, 0, stream>>>(Wq, WtQ);
    wtrans_kernel<<<(wtot + 255) / 256, 256, 0, stream>>>(Wk, WtK);
    wtrans_kernel<<<(wtot + 255) / 256, 256, 0, stream>>>(Wv, WtV);

    // QKV projection + symmetric fake-quant (int codes stored in f16)
    // strips = (B*S/16) * (HID/64) = 512*12 = 6144 waves -> 768 blocks
    qproj_kernel<<<768, 256, 0, stream>>>(hs, WtQ, bq, s_q, Qh, 0);
    qproj_kernel<<<768, 256, 0, stream>>>(hs, WtK, bk, s_k, Kh, 0);
    qproj_kernel<<<768, 256, 0, stream>>>(hs, WtV, bv, s_v, VT, 1);

    // Attention: B*NH*(S/16) = 6144 waves -> 768 blocks of 8 waves
    qattn_kernel<<<768, 256, 0, stream>>>(Qh, Kh, VT, mask,
                                          s_q, s_k, s_v, s_attn, s_after,
                                          (float*)d_out);
}
